// Attention_48816598286380
// MI455X (gfx1250) — matrix operand order repaired and checked by reading, not compiled, not run on hardware
//
#include <hip/hip_runtime.h>

typedef __attribute__((ext_vector_type(16))) _Float16 v16h;
typedef __attribute__((ext_vector_type(8)))  _Float16 v8h;
typedef __attribute__((ext_vector_type(8)))  float    v8f;
typedef __attribute__((ext_vector_type(4)))  float    v4f;

#define B_    4
#define N_    512
#define M_    8192
#define QDIM  512
#define IDIM  256
#define HEADS 8
#define DHEAD 64
#define ATT   512   // HEADS*DHEAD

// ---------------- WMMA helpers (gfx1250, wave32) ----------------

__device__ __forceinline__ v8f wmma16x16x32(v16h a, v16h b, v8f c) {
  // (neg_a, A, neg_b, B, c_mod, C, reuse_a, reuse_b)
  return __builtin_amdgcn_wmma_f32_16x16x32_f16(false, a, false, b, (short)0, c,
                                                false, false);
}

// A fragment: 16(M) x 32(K). A row-major, leading dim lda (halves). p -> &A[m0][k0].
// lane l: row l%16; element e = K (l/16)*16 + e.
__device__ __forceinline__ v16h load_frag_a(const _Float16* p, int lda) {
  const int lane = threadIdx.x & 31;
  const _Float16* r0 = p + (lane & 15) * lda + ((lane >> 4) << 4);
  v8h lo = *(const v8h*)(r0);
  v8h hi = *(const v8h*)(r0 + 8);
  v16h r;
#pragma unroll
  for (int i = 0; i < 8; ++i) { r[i] = lo[i]; r[i + 8] = hi[i]; }
  return r;
}

// B fragment: 32(K) x 16(N) from Bt = B^T stored N x K row-major. p -> &Bt[n0][k0].
// lane l: column l%16; element e = K (l/16)*16 + e.  One 32B contiguous load.
__device__ __forceinline__ v16h load_frag_b(const _Float16* p, int ldb) {
  const int lane = threadIdx.x & 31;
  const _Float16* r0 = p + (lane & 15) * ldb + ((lane >> 4) << 4);
  return *(const v16h*)(r0);
}

// 32(M) x 64(N) tile, software-pipelined over K (K % 32 == 0).
// acc[g*4+f]: row-group g (rows m0+g*16..), col-group f (n0+f*16..)
__device__ __forceinline__ void gemm_acc_32x64(const _Float16* A, int lda,
                                               const _Float16* Bt, int ldb,
                                               int K, v8f acc[8]) {
  v16h a0 = load_frag_a(A, lda);
  v16h a1 = load_frag_a(A + 16 * lda, lda);
  v16h b0 = load_frag_b(Bt + 0 * ldb, ldb);
  v16h b1 = load_frag_b(Bt + 16 * ldb, ldb);
  v16h b2 = load_frag_b(Bt + 32 * ldb, ldb);
  v16h b3 = load_frag_b(Bt + 48 * ldb, ldb);
  for (int k0 = 0; k0 < K; k0 += 32) {
    v16h a0c = a0, a1c = a1, b0c = b0, b1c = b1, b2c = b2, b3c = b3;
    // prefetch next k-step (wraps to 0 on last iter; redundant loads are cheap
    // and keep the loop branchless/uniform)
    int kn = (k0 + 32 < K) ? (k0 + 32) : 0;
    a0 = load_frag_a(A + kn, lda);
    a1 = load_frag_a(A + 16 * lda + kn, lda);
    b0 = load_frag_b(Bt + 0 * ldb + kn, ldb);
    b1 = load_frag_b(Bt + 16 * ldb + kn, ldb);
    b2 = load_frag_b(Bt + 32 * ldb + kn, ldb);
    b3 = load_frag_b(Bt + 48 * ldb + kn, ldb);
    acc[0] = wmma16x16x32(a0c, b0c, acc[0]);
    acc[1] = wmma16x16x32(a0c, b1c, acc[1]);
    acc[2] = wmma16x16x32(a0c, b2c, acc[2]);
    acc[3] = wmma16x16x32(a0c, b3c, acc[3]);
    acc[4] = wmma16x16x32(a1c, b0c, acc[4]);
    acc[5] = wmma16x16x32(a1c, b1c, acc[5]);
    acc[6] = wmma16x16x32(a1c, b2c, acc[6]);
    acc[7] = wmma16x16x32(a1c, b3c, acc[7]);
  }
}

// ---------------- data-prep kernels ----------------

__global__ void cvt_f32_to_f16x4(const float* __restrict__ src,
                                 _Float16* __restrict__ dst, int n4) {
  int i = blockIdx.x * blockDim.x + threadIdx.x;
  if (i < n4) {
    v4f v = *(const v4f*)(src + i * 4);
    _Float16* d = dst + i * 4;
    d[0] = (_Float16)v[0]; d[1] = (_Float16)v[1];
    d[2] = (_Float16)v[2]; d[3] = (_Float16)v[3];
  }
}

// W: [K][N] row-major fp32 -> Wt: [N][K] row-major f16
__global__ void transpose_w(const float* __restrict__ W,
                            _Float16* __restrict__ Wt, int K, int N) {
  int i = blockIdx.x * blockDim.x + threadIdx.x;
  if (i < K * N) {
    int k = i / N, n = i - k * N;
    Wt[n * K + k] = (_Float16)W[i];
  }
}

// ---------------- projection kernels (one wave -> 32x64 C tile) ----------------

// q = x @ Wq, scaled by 1/8, written as qh[(b*8+h)][n][d]
__global__ void __launch_bounds__(128) qproj_kernel(
    const _Float16* __restrict__ xh, const _Float16* __restrict__ WqT,
    _Float16* __restrict__ qh) {
  const int lane = threadIdx.x & 31;
  const int gw = (blockIdx.x * blockDim.x + threadIdx.x) >> 5;
  const int tilesN = ATT / 64;                 // 8
  const int m0 = (gw / tilesN) * 32;           // rows over B*N = 2048
  const int n0 = (gw % tilesN) * 64;
  v8f z = {0, 0, 0, 0, 0, 0, 0, 0};
  v8f acc[8] = {z, z, z, z, z, z, z, z};
  gemm_acc_32x64(xh + (size_t)m0 * QDIM, QDIM, WqT + (size_t)n0 * QDIM, QDIM,
                 QDIM, acc);
#pragma unroll
  for (int g = 0; g < 2; ++g) {
    int rbase = m0 + g * 16 + ((lane >> 4) << 3);
#pragma unroll
    for (int r = 0; r < 8; ++r) {
      int row = rbase + r;
      int b = row >> 9, n = row & 511;
#pragma unroll
      for (int f = 0; f < 4; ++f) {
        int col = n0 + f * 16 + (lane & 15);
        int h = col >> 6, d = col & 63;
        qh[(((b << 3) + h) * (size_t)N_ + n) * DHEAD + d] =
            (_Float16)(acc[g * 4 + f][r] * 0.125f);
      }
    }
  }
}

// kv = context @ Wkv; k -> kh[(b*8+h)][m][d]; v -> vT[(b*8+h)][d][m]
__global__ void __launch_bounds__(128) kvproj_kernel(
    const _Float16* __restrict__ ch, const _Float16* __restrict__ WkvT,
    _Float16* __restrict__ kh, _Float16* __restrict__ vT) {
  const int lane = threadIdx.x & 31;
  const int gw = (blockIdx.x * blockDim.x + threadIdx.x) >> 5;
  const int tilesN = (2 * ATT) / 64;           // 16
  const int m0 = (gw / tilesN) * 32;           // rows over B*M = 32768
  const int n0 = (gw % tilesN) * 64;
  v8f z = {0, 0, 0, 0, 0, 0, 0, 0};
  v8f acc[8] = {z, z, z, z, z, z, z, z};
  gemm_acc_32x64(ch + (size_t)m0 * IDIM, IDIM, WkvT + (size_t)n0 * IDIM, IDIM,
                 IDIM, acc);
#pragma unroll
  for (int g = 0; g < 2; ++g) {
    int rbase = m0 + g * 16 + ((lane >> 4) << 3);
#pragma unroll
    for (int r = 0; r < 8; ++r) {
      int row = rbase + r;
      int b = row >> 13, m = row & 8191;
#pragma unroll
      for (int f = 0; f < 4; ++f) {
        int col = n0 + f * 16 + (lane & 15);
        _Float16 val = (_Float16)acc[g * 4 + f][r];
        if (col < ATT) {
          int h = col >> 6, d = col & 63;
          kh[(((b << 3) + h) * (size_t)M_ + m) * DHEAD + d] = val;
        } else {
          int c2 = col - ATT;
          int h = c2 >> 6, d = c2 & 63;
          vT[((((b << 3) + h) * (size_t)DHEAD) + d) * M_ + m] = val;
        }
      }
    }
  }
}

// out = oh @ Wo + bo  (f32 output)
__global__ void __launch_bounds__(128) oproj_kernel(
    const _Float16* __restrict__ oh, const _Float16* __restrict__ WoT,
    const float* __restrict__ bo, float* __restrict__ out) {
  const int lane = threadIdx.x & 31;
  const int gw = (blockIdx.x * blockDim.x + threadIdx.x) >> 5;
  const int tilesN = QDIM / 64;                // 8
  const int m0 = (gw / tilesN) * 32;           // rows over B*N = 2048
  const int n0 = (gw % tilesN) * 64;
  v8f z = {0, 0, 0, 0, 0, 0, 0, 0};
  v8f acc[8] = {z, z, z, z, z, z, z, z};
  gemm_acc_32x64(oh + (size_t)m0 * ATT, ATT, WoT + (size_t)n0 * QDIM, QDIM,
                 ATT, acc);
#pragma unroll
  for (int g = 0; g < 2; ++g) {
    int rbase = m0 + g * 16 + ((lane >> 4) << 3);
#pragma unroll
    for (int r = 0; r < 8; ++r) {
      int row = rbase + r;
#pragma unroll
      for (int f = 0; f < 4; ++f) {
        int col = n0 + f * 16 + (lane & 15);
        out[(size_t)row * QDIM + col] = acc[g * 4 + f][r] + bo[col];
      }
    }
  }
}

// ---------------- flash attention (one wave -> 32 queries x full head) ----------------

__global__ void __launch_bounds__(128) attn_kernel(
    const _Float16* __restrict__ qh, const _Float16* __restrict__ kh,
    const _Float16* __restrict__ vT, _Float16* __restrict__ oh) {
  __shared__ __align__(64) _Float16 Plds[4][2 * 16 * 32];
  const int wave = threadIdx.x >> 5;
  const int lane = threadIdx.x & 31;
  const int gw = blockIdx.x * 4 + wave;        // 0..511
  const int qt = gw & 15;                      // 32-query tile within head
  const int bh = gw >> 4;                      // 0..31 (b*8+h)

  const _Float16* Q  = qh + ((size_t)bh * N_ + qt * 32) * DHEAD;
  const _Float16* Kp = kh + (size_t)bh * M_ * DHEAD;
  const _Float16* Vt = vT + (size_t)bh * DHEAD * M_;
  _Float16* P = &Plds[wave][0];

  // Q fragments: 2 query sub-tiles x 2 k-halves (loaded once)
  v16h qa00 = load_frag_a(Q, DHEAD);
  v16h qa01 = load_frag_a(Q + 32, DHEAD);
  v16h qa10 = load_frag_a(Q + 16 * DHEAD, DHEAD);
  v16h qa11 = load_frag_a(Q + 16 * DHEAD + 32, DHEAD);

  float mrow[2][8], lrow[2][8];
  v8f z = {0, 0, 0, 0, 0, 0, 0, 0};
  v8f acc[2][4] = {{z, z, z, z}, {z, z, z, z}};
#pragma unroll
  for (int t = 0; t < 2; ++t)
#pragma unroll
    for (int r = 0; r < 8; ++r) { mrow[t][r] = -1e30f; lrow[t][r] = 0.0f; }

  // prefetch K fragments for chunk 0
  v16h kb0 = load_frag_b(Kp + 0, DHEAD);
  v16h kb1 = load_frag_b(Kp + 32, DHEAD);
  v16h kb2 = load_frag_b(Kp + 16 * DHEAD, DHEAD);
  v16h kb3 = load_frag_b(Kp + 16 * DHEAD + 32, DHEAD);

  for (int j0 = 0; j0 < M_; j0 += 32) {
    // issue V loads early: in flight through the score WMMAs + softmax
    v16h vb0 = load_frag_b(Vt + (size_t)0  * M_ + j0, M_);
    v16h vb1 = load_frag_b(Vt + (size_t)16 * M_ + j0, M_);
    v16h vb2 = load_frag_b(Vt + (size_t)32 * M_ + j0, M_);
    v16h vb3 = load_frag_b(Vt + (size_t)48 * M_ + j0, M_);

    // scores: S[t][j-subtile], each 16x16 over 32 keys
    v8f s00 = z, s01 = z, s10 = z, s11 = z;
    s00 = wmma16x16x32(qa00, kb0, s00); s00 = wmma16x16x32(qa01, kb1, s00);
    s01 = wmma16x16x32(qa00, kb2, s01); s01 = wmma16x16x32(qa01, kb3, s01);
    s10 = wmma16x16x32(qa10, kb0, s10); s10 = wmma16x16x32(qa11, kb1, s10);
    s11 = wmma16x16x32(qa10, kb2, s11); s11 = wmma16x16x32(qa11, kb3, s11);

    // prefetch next chunk's K fragments (wrap to 0 on last iter, unused)
    {
      int jn = (j0 + 32 < M_) ? (j0 + 32) : 0;
      kb0 = load_frag_b(Kp + (size_t)jn * DHEAD, DHEAD);
      kb1 = load_frag_b(Kp + (size_t)jn * DHEAD + 32, DHEAD);
      kb2 = load_frag_b(Kp + (size_t)(jn + 16) * DHEAD, DHEAD);
      kb3 = load_frag_b(Kp + (size_t)(jn + 16) * DHEAD + 32, DHEAD);
    }

    // online softmax per row (each row lives in one 16-lane half)
#pragma unroll
    for (int t = 0; t < 2; ++t) {
      v8f sA = t ? s10 : s00;
      v8f sB = t ? s11 : s01;
      _Float16* Pt = P + t * 512;
#pragma unroll
      for (int r = 0; r < 8; ++r) {
        float mx = fmaxf(sA[r], sB[r]);
#pragma unroll
        for (int off = 1; off < 16; off <<= 1)
          mx = fmaxf(mx, __shfl_xor(mx, off, 32));
        float mnew = fmaxf(mrow[t][r], mx);
        float p0 = __expf(sA[r] - mnew);
        float p1 = __expf(sB[r] - mnew);
        float rs = p0 + p1;
#pragma unroll
        for (int off = 1; off < 16; off <<= 1) rs += __shfl_xor(rs, off, 32);
        float c = __expf(mrow[t][r] - mnew);
        lrow[t][r] = lrow[t][r] * c + rs;
        mrow[t][r] = mnew;
#pragma unroll
        for (int f = 0; f < 4; ++f) acc[t][f][r] *= c;
        int mm = r + ((lane >> 4) << 3);
        Pt[mm * 32 + (lane & 15)]      = (_Float16)p0;
        Pt[mm * 32 + 16 + (lane & 15)] = (_Float16)p1;
      }
    }
    asm volatile("s_wait_dscnt 0" ::: "memory");

    // O[t] += P[t][16x32] @ V[32x64]
    v16h pa0 = load_frag_a(P, 32);
    v16h pa1 = load_frag_a(P + 512, 32);
    acc[0][0] = wmma16x16x32(pa0, vb0, acc[0][0]);
    acc[0][1] = wmma16x16x32(pa0, vb1, acc[0][1]);
    acc[0][2] = wmma16x16x32(pa0, vb2, acc[0][2]);
    acc[0][3] = wmma16x16x32(pa0, vb3, acc[0][3]);
    acc[1][0] = wmma16x16x32(pa1, vb0, acc[1][0]);
    acc[1][1] = wmma16x16x32(pa1, vb1, acc[1][1]);
    acc[1][2] = wmma16x16x32(pa1, vb2, acc[1][2]);
    acc[1][3] = wmma16x16x32(pa1, vb3, acc[1][3]);
  }

  // epilogue: normalize and write oh[b][n][h*64+d]
  const int b = bh >> 3, h = bh & 7;
#pragma unroll
  for (int t = 0; t < 2; ++t) {
#pragma unroll
    for (int r = 0; r < 8; ++r) {
      float inv = 1.0f / lrow[t][r];
      int n = qt * 32 + t * 16 + r + ((lane >> 4) << 3);
#pragma unroll
      for (int f = 0; f < 4; ++f) {
        int d = f * 16 + (lane & 15);
        oh[((size_t)(b * N_ + n)) * ATT + h * DHEAD + d] =
            (_Float16)(acc[t][f][r] * inv);
      }
    }
  }
}

// ---------------- host launcher ----------------

extern "C" void kernel_launch(void* const* d_in, const int* in_sizes, int n_in,
                              void* d_out, int out_size, void* d_ws, size_t ws_size,
                              hipStream_t stream) {
  const float* x   = (const float*)d_in[0];  // [B,N,QDIM]
  const float* ctx = (const float*)d_in[1];  // [B,M,IDIM]
  const float* Wq  = (const float*)d_in[2];  // [QDIM,ATT]
  const float* Wkv = (const float*)d_in[3];  // [IDIM,2*ATT]
  const float* Wo  = (const float*)d_in[4];  // [ATT,QDIM]
  const float* bo  = (const float*)d_in[5];  // [QDIM]
  float* out = (float*)d_out;

  size_t off = 0;
  auto suballoc = [&](size_t bytes) -> char* {
    char* p = (char*)d_ws + off;
    off += (bytes + 255) & ~(size_t)255;
    return p;
  };
  const size_t nx = (size_t)B_ * N_ * QDIM;          // 1,048,576
  const size_t nc = (size_t)B_ * M_ * IDIM;          // 8,388,608
  _Float16* xh   = (_Float16*)suballoc(nx * 2);
  _Float16* chh  = (_Float16*)suballoc(nc * 2);
  _Float16* WqT  = (_Float16*)suballoc((size_t)QDIM * ATT * 2);
  _Float16* WkvT = (_Float16*)suballoc((size_t)IDIM * 2 * ATT * 2);
  _Float16* WoT  = (_Float16*)suballoc((size_t)ATT * QDIM * 2);
  _Float16* qh   = (_Float16*)suballoc((size_t)B_ * HEADS * N_ * DHEAD * 2);
  _Float16* kh   = (_Float16*)suballoc((size_t)B_ * HEADS * M_ * DHEAD * 2);
  _Float16* vT   = (_Float16*)suballoc((size_t)B_ * HEADS * DHEAD * M_ * 2);
  _Float16* oh   = (_Float16*)suballoc((size_t)B_ * N_ * ATT * 2);
  (void)ws_size; (void)in_sizes; (void)n_in; (void)out_size;

  // 1. fp32 -> f16 conversions (x4 vectorized)
  cvt_f32_to_f16x4<<<(int)(nx / 4 / 256), 256, 0, stream>>>(x, xh, (int)(nx / 4));
  cvt_f32_to_f16x4<<<(int)(nc / 4 / 256), 256, 0, stream>>>(ctx, chh, (int)(nc / 4));
  transpose_w<<<(QDIM * ATT + 255) / 256, 256, 0, stream>>>(Wq, WqT, QDIM, ATT);
  transpose_w<<<(IDIM * 2 * ATT + 255) / 256, 256, 0, stream>>>(Wkv, WkvT, IDIM, 2 * ATT);
  transpose_w<<<(ATT * QDIM + 255) / 256, 256, 0, stream>>>(Wo, WoT, ATT, QDIM);

  // 2. projections (one wave per 32x64 C tile, 4 waves per block)
  // qproj: (2048/32)*(512/64) = 512 waves -> 128 blocks
  qproj_kernel<<<128, 128, 0, stream>>>(xh, WqT, qh);
  // kvproj: (32768/32)*(1024/64) = 16384 waves -> 4096 blocks
  kvproj_kernel<<<4096, 128, 0, stream>>>(chh, WkvT, kh, vT);

  // 3. flash attention: 32 heads * 16 q-tiles(32q) = 512 waves -> 128 blocks
  attn_kernel<<<128, 128, 0, stream>>>(qh, kh, vT, oh);

  // 4. output projection (+bias): 512 waves -> 128 blocks
  oproj_kernel<<<128, 128, 0, stream>>>(oh, WoT, bo, out);
}
